// LSTMBranch_5291399708735
// MI455X (gfx1250) — matrix operand order repaired and checked by reading, not compiled, not run on hardware
//
#include <hip/hip_runtime.h>
#include <stdint.h>

#define B_  256
#define T_  512
#define DIN 64
#define H_  128
#define G4H 512   // 4*H
#define H2  256   // 2*H

typedef __bf16 bf16x16 __attribute__((ext_vector_type(16)));
typedef float  f32x8   __attribute__((ext_vector_type(8)));

union Frag  { uint4 q[2]; uint32_t u[8]; bf16x16 v; };  // 16 bf16 (A or B operand)
union CFrag { float f[8]; f32x8 v; };                    // 8 f32 (C/D accumulator)

// ---- CDNA5 async global->LDS path (guarded; falls back to direct loads) ----
#if __has_builtin(__builtin_amdgcn_global_load_async_to_lds_b128)
#define HAVE_ASYNC 1
typedef int v4i_ __attribute__((ext_vector_type(4)));
typedef __attribute__((address_space(1))) v4i_* gp4;
typedef __attribute__((address_space(3))) v4i_* lp4;
__device__ __forceinline__ void async_cp16(const uint4* g, uint4* l) {
    __builtin_amdgcn_global_load_async_to_lds_b128((gp4)const_cast<uint4*>(g), (lp4)l, 0, 0);
}
#if __has_builtin(__builtin_amdgcn_s_wait_asynccnt)
#define WAIT_ASYNC(n) __builtin_amdgcn_s_wait_asynccnt(n)
#else
#define WAIT_ASYNC(n) asm volatile("s_wait_asynccnt %0" ::"n"(n))
#endif
#endif

__device__ __forceinline__ uint16_t f2bf(float f) {
    uint32_t x = __float_as_uint(f);
    x += 0x7FFFu + ((x >> 16) & 1u);          // round-to-nearest-even
    return (uint16_t)(x >> 16);
}

// tanh: gfx1250 has V_TANH_F32 (1 trans op); fall back to exp+rcp if builtin absent
__device__ __forceinline__ float t_tanh(float x) {
#if __has_builtin(__builtin_amdgcn_tanhf)
    return __builtin_amdgcn_tanhf(x);
#else
    float e = __expf(-2.0f * x);
    float d = 1.0f + e;
#if __has_builtin(__builtin_amdgcn_rcpf)
    float r = __builtin_amdgcn_rcpf(d);
#else
    float r = 1.0f / d;
#endif
    return fmaf(2.0f, r, -1.0f);
#endif
}
__device__ __forceinline__ float t_sigm(float x) { return fmaf(0.5f, t_tanh(0.5f * x), 0.5f); }

// ---------------------------------------------------------------- helpers ----
__global__ void cvt_f32_bf16(const float* __restrict__ src, uint16_t* __restrict__ dst, int n) {
    int i = blockIdx.x * blockDim.x + threadIdx.x;
    if (i < n) dst[i] = f2bf(src[i]);
}
__global__ void bias_sum(const float* __restrict__ a, const float* __restrict__ b,
                         float* __restrict__ o, int n) {
    int i = blockIdx.x * blockDim.x + threadIdx.x;
    if (i < n) o[i] = a[i] + b[i];
}

// fragment-ordered pre-activations: uint4 index [dir][t][bt(16)][jt(32)][lane(32)],
// each uint4 = lane's 8 C-fragment values (rows M = r + 8*(lane/16), col N = lane%16) in bf16
__device__ __forceinline__ size_t pre_idx(int dir, int t, int bt, int jt, int lane) {
    return ((((size_t)dir * T_ + t) * 16 + bt) * 32 + jt) * 32 + lane;
}

// ------------------------------------------------- input projection (GEMM) ----
// pre[dir][t][b][4H] = xin[b][t][:] @ Wih[dir]^T + bias[dir], emitted in fragment order
template <int KIN>
__global__ __launch_bounds__(256) void proj_kernel(
    const uint16_t* __restrict__ xin,   // [B][T][KIN]  bf16 (row-major, K contiguous)
    const uint16_t* __restrict__ wih,   // [2][4H][KIN] bf16
    const float*    __restrict__ bias,  // [2][4H]      f32 (bih+bhh)
    uint4*          __restrict__ pre4)  // fragment-ordered bf16
{
    const int lane = threadIdx.x & 31;
    const int wave = threadIdx.x >> 5;
    const int g    = lane >> 4;
    const int nn   = lane & 15;

    const int m0  = blockIdx.x * 16;          // tile over M = T*B (row m = t*B + b)
    const int jt  = blockIdx.y * 8 + wave;    // 0..31
    const int j0  = jt * 16;
    const int dir = blockIdx.z;

    const int t  = m0 / B_;
    const int b0 = m0 % B_;

    const uint16_t* wd = wih + (size_t)dir * G4H * KIN;

    // A fragments: 16-bit A 16x32 layout -> two contiguous 16B runs per fragment
    Frag A[KIN / 32];
    const uint16_t* arow = xin + ((size_t)(b0 + (lane & 15)) * T_ + t) * KIN;
#pragma unroll
    for (int kc = 0; kc < KIN / 32; ++kc) {
        A[kc].q[0] = *(const uint4*)(arow + kc * 32 + 8 * g);
        A[kc].q[1] = *(const uint4*)(arow + kc * 32 + 8 * g + 16);
    }

    CFrag acc;
    const float bv = bias[dir * G4H + j0 + nn];
#pragma unroll
    for (int r = 0; r < 8; ++r) acc.f[r] = bv;

#pragma unroll
    for (int kc = 0; kc < KIN / 32; ++kc) {
        Frag Bf;  // B 32x16: lane group g holds K=8g..8g+7 and 16+8g..16+8g+7, col = j0+nn
        const uint16_t* bcol = wd + (size_t)(j0 + nn) * KIN + kc * 32 + 8 * g;
        Bf.q[0] = *(const uint4*)(bcol);
        Bf.q[1] = *(const uint4*)(bcol + 16);
        acc.v = __builtin_amdgcn_wmma_f32_16x16x32_bf16(
            false, A[kc].v, false, Bf.v, (short)0, acc.v, false, false);
    }

    // pack 8 f32 -> 8 bf16 and store one uint4 (coalesced, fragment order)
    uint4 st;
    st.x = (uint32_t)f2bf(acc.f[0]) | ((uint32_t)f2bf(acc.f[1]) << 16);
    st.y = (uint32_t)f2bf(acc.f[2]) | ((uint32_t)f2bf(acc.f[3]) << 16);
    st.z = (uint32_t)f2bf(acc.f[4]) | ((uint32_t)f2bf(acc.f[5]) << 16);
    st.w = (uint32_t)f2bf(acc.f[6]) | ((uint32_t)f2bf(acc.f[7]) << 16);
    pre4[pre_idx(dir, t, b0 >> 4, jt, lane)] = st;
}

// ------------------------------------------------------------ LSTM scan ------
// grid = (2 dirs, B/16), block = 256 threads (8 waves; wave = k-tile kt).
// Whh resident in 128 VGPRs/lane, c (and time-mean) in registers, h ping-pong in LDS.
// Pre-activations for step t+1 streamed global->LDS asynchronously during step t;
// each wave stages exactly the tiles it consumes (no extra barrier, drained by asynccnt).
template <int FINAL>
__global__ __launch_bounds__(256) void scan_kernel(
    const uint4*    __restrict__ pre4,  // fragment-ordered [dir][t][bt][jt][lane]
    const uint16_t* __restrict__ whh,   // [2][4H][H] bf16
    uint16_t*       __restrict__ h1,    // [B][T][2H] bf16 (layer-0 out) or null
    float*          __restrict__ out)   // [B][2H]    f32  (time mean)   or null
{
    __shared__ uint16_t h_lds[2][16][H_];
#ifdef HAVE_ASYNC
    __shared__ uint4 pre_s[2][32][32];  // [buf][jt][lane], 32 KB double buffer
#endif

    const int tid  = threadIdx.x;
    const int lane = tid & 31;
    const int kt   = tid >> 5;        // wave id == k-tile 0..7
    const int g    = lane >> 4;
    const int nn   = lane & 15;
    const int dir  = blockIdx.x;
    const int bt   = blockIdx.y;
    const int b0   = bt * 16;

    // ---- load all 16 recurrent-weight fragments into registers (constant over T) ----
    Frag Breg[16];
    const uint16_t* wg = whh + (size_t)dir * G4H * H_;
#pragma unroll
    for (int gate = 0; gate < 4; ++gate)
#pragma unroll
        for (int kc = 0; kc < 4; ++kc) {
            const uint16_t* p = wg + (size_t)(gate * H_ + kt * 16 + nn) * H_ + kc * 32 + 8 * g;
            Breg[gate * 4 + kc].q[0] = *(const uint4*)(p);
            Breg[gate * 4 + kc].q[1] = *(const uint4*)(p + 16);
        }

#ifdef HAVE_ASYNC
    { // stage step 0 pre-activations (this wave's 4 tiles)
        const int t0 = dir ? (T_ - 1) : 0;
#pragma unroll
        for (int gate = 0; gate < 4; ++gate)
            async_cp16(&pre4[pre_idx(dir, t0, bt, gate * 8 + kt, lane)],
                       &pre_s[0][gate * 8 + kt][lane]);
    }
#endif

    { // zero h ping-pong
        uint32_t* hz = (uint32_t*)h_lds;
        for (int i = tid; i < 2 * 16 * H_ / 2; i += 256) hz[i] = 0u;
    }
    __syncthreads();

    float c_frag[8], hsum[8];
#pragma unroll
    for (int r = 0; r < 8; ++r) { c_frag[r] = 0.f; hsum[r] = 0.f; }

    for (int step = 0; step < T_; ++step) {
        const int t   = dir ? (T_ - 1 - step) : step;
        const int cur = step & 1, prv = cur ^ 1;

        uint4 pv[4];
#ifdef HAVE_ASYNC
        if (step + 1 < T_) {  // stage t+1 while we compute t, then drain t's batch
            const int tn = dir ? (T_ - 2 - step) : step + 1;
#pragma unroll
            for (int gate = 0; gate < 4; ++gate)
                async_cp16(&pre4[pre_idx(dir, tn, bt, gate * 8 + kt, lane)],
                           &pre_s[prv][gate * 8 + kt][lane]);
            WAIT_ASYNC(4);   // async loads complete in order: only the 4 newest remain
        } else {
            WAIT_ASYNC(0);
        }
#pragma unroll
        for (int gate = 0; gate < 4; ++gate)
            pv[gate] = pre_s[cur][gate * 8 + kt][lane];
#else
#pragma unroll
        for (int gate = 0; gate < 4; ++gate)
            pv[gate] = pre4[pre_idx(dir, t, bt, gate * 8 + kt, lane)];
        if (step + 1 < T_) {
            const int tn = dir ? (T_ - 2 - step) : step + 1;
            __builtin_prefetch((const void*)&pre4[pre_idx(dir, tn, bt, kt, lane)], 0, 0);
        }
#endif

        // A = h[t-1] (16 x 128) from LDS: two b128 per K-chunk
        Frag A[4];
        const uint16_t* hrow = &h_lds[prv][lane & 15][0];
#pragma unroll
        for (int kc = 0; kc < 4; ++kc) {
            A[kc].q[0] = *(const uint4*)(hrow + kc * 32 + 8 * g);
            A[kc].q[1] = *(const uint4*)(hrow + kc * 32 + 8 * g + 16);
        }

        CFrag gacc[4];
#pragma unroll
        for (int gate = 0; gate < 4; ++gate) {
            const uint32_t* pu = (const uint32_t*)&pv[gate];
#pragma unroll
            for (int i = 0; i < 4; ++i) {   // unpack bf16 pair -> two f32 accum inits
                uint32_t u = pu[i];
                gacc[gate].f[2 * i]     = __uint_as_float(u << 16);
                gacc[gate].f[2 * i + 1] = __uint_as_float(u & 0xFFFF0000u);
            }
#pragma unroll
            for (int kc = 0; kc < 4; ++kc)
                gacc[gate].v = __builtin_amdgcn_wmma_f32_16x16x32_bf16(
                    false, A[kc].v, false, Breg[gate * 4 + kc].v, (short)0,
                    gacc[gate].v, false, false);
        }

#pragma unroll
        for (int r = 0; r < 8; ++r) {     // gates (i,f,g,o) in f32; 1 trans op each
            float iv = t_sigm(gacc[0].f[r]);
            float fv = t_sigm(gacc[1].f[r]);
            float gv = t_tanh(gacc[2].f[r]);
            float ov = t_sigm(gacc[3].f[r]);
            float c  = fmaf(fv, c_frag[r], iv * gv);
            c_frag[r] = c;
            float h  = ov * t_tanh(c);
            if (FINAL) hsum[r] += h;
            h_lds[cur][r + 8 * g][kt * 16 + nn] = f2bf(h);
        }
        __syncthreads();

        if (!FINAL) {  // coalesced b128 store of the fresh h tile to global [b][t][2H]
            const int row = tid >> 4;
            const int c0  = (tid & 15) * 8;
            uint4 hv = *(const uint4*)&h_lds[cur][row][c0];
            *(uint4*)&h1[((size_t)(b0 + row) * T_ + t) * H2 + dir * H_ + c0] = hv;
        }
    }

    if (FINAL) {
        const float inv = 1.0f / (float)T_;
#pragma unroll
        for (int r = 0; r < 8; ++r)
            out[(size_t)(b0 + r + 8 * g) * H2 + dir * H_ + kt * 16 + nn] = hsum[r] * inv;
    }
}

// ------------------------------------------------------------------ host -----
extern "C" void kernel_launch(void* const* d_in, const int* in_sizes, int n_in,
                              void* d_out, int out_size, void* d_ws, size_t ws_size,
                              hipStream_t stream) {
    (void)in_sizes; (void)n_in; (void)out_size; (void)ws_size;
    const float* x     = (const float*)d_in[0];
    const float* wih0f = (const float*)d_in[1];  const float* whh0f = (const float*)d_in[2];
    const float* bih0f = (const float*)d_in[3];  const float* bhh0f = (const float*)d_in[4];
    const float* wih0b = (const float*)d_in[5];  const float* whh0b = (const float*)d_in[6];
    const float* bih0b = (const float*)d_in[7];  const float* bhh0b = (const float*)d_in[8];
    const float* wih1f = (const float*)d_in[9];  const float* whh1f = (const float*)d_in[10];
    const float* bih1f = (const float*)d_in[11]; const float* bhh1f = (const float*)d_in[12];
    const float* wih1b = (const float*)d_in[13]; const float* whh1b = (const float*)d_in[14];
    const float* bih1b = (const float*)d_in[15]; const float* bhh1b = (const float*)d_in[16];

    uint8_t* ws = (uint8_t*)d_ws;
    size_t off = 0;
    auto alloc = [&](size_t bytes) -> uint8_t* {
        uint8_t* p = ws + off; off += (bytes + 255) & ~(size_t)255; return p;
    };
    uint16_t* xbf  = (uint16_t*)alloc((size_t)B_ * T_ * DIN * 2);
    uint16_t* wih0 = (uint16_t*)alloc((size_t)2 * G4H * DIN * 2);
    uint16_t* wih1 = (uint16_t*)alloc((size_t)2 * G4H * H2 * 2);
    uint16_t* whhb = (uint16_t*)alloc((size_t)4 * G4H * H_ * 2);   // [layer][dir]
    float*    bias = (float*)   alloc((size_t)4 * G4H * 4);        // [layer][dir]
    uint16_t* h1   = (uint16_t*)alloc((size_t)B_ * T_ * H2 * 2);
    uint4*    pre4 = (uint4*)   alloc((size_t)2 * T_ * B_ * G4H * 2);

    auto cvt = [&](const float* s, uint16_t* d, int n) {
        cvt_f32_bf16<<<(n + 255) / 256, 256, 0, stream>>>(s, d, n);
    };
    cvt(x, xbf, B_ * T_ * DIN);
    cvt(wih0f, wih0,              G4H * DIN);
    cvt(wih0b, wih0 + G4H * DIN,  G4H * DIN);
    cvt(wih1f, wih1,              G4H * H2);
    cvt(wih1b, wih1 + G4H * H2,   G4H * H2);
    cvt(whh0f, whhb + 0 * G4H * H_, G4H * H_);
    cvt(whh0b, whhb + 1 * G4H * H_, G4H * H_);
    cvt(whh1f, whhb + 2 * G4H * H_, G4H * H_);
    cvt(whh1b, whhb + 3 * G4H * H_, G4H * H_);
    bias_sum<<<2, 256, 0, stream>>>(bih0f, bhh0f, bias + 0 * G4H, G4H);
    bias_sum<<<2, 256, 0, stream>>>(bih0b, bhh0b, bias + 1 * G4H, G4H);
    bias_sum<<<2, 256, 0, stream>>>(bih1f, bhh1f, bias + 2 * G4H, G4H);
    bias_sum<<<2, 256, 0, stream>>>(bih1b, bhh1b, bias + 3 * G4H, G4H);

    const dim3 pgrid(T_ * B_ / 16, 4, 2);
    proj_kernel<DIN><<<pgrid, 256, 0, stream>>>(xbf, wih0, bias, pre4);
    scan_kernel<0><<<dim3(2, B_ / 16), 256, 0, stream>>>(pre4, whhb, h1, nullptr);
    proj_kernel<H2><<<pgrid, 256, 0, stream>>>(h1, wih1, bias + 2 * G4H, pre4);
    scan_kernel<1><<<dim3(2, B_ / 16), 256, 0, stream>>>(pre4, whhb + 2 * G4H * H_,
                                                         nullptr, (float*)d_out);
}
